// DiTBlock_33775622816508
// MI455X (gfx1250) — matrix operand order repaired and checked by reading, not compiled, not run on hardware
//
#include <hip/hip_runtime.h>
#include <hip/hip_bf16.h>

// ---------------------------------------------------------------------------
// DiT block for MI455X (gfx1250, wave32, WMMA).
// fp32 in/out; GEMMs in f16 via v_wmma_f32_16x16x32_f16 with fp32 accumulate.
// Weights are transposed once (W^T [N][K]) so ALL wmma fragments are
// contiguous 32-byte LDS reads (ds_load_b128 pairs), no scalar gathers.
// ---------------------------------------------------------------------------

typedef __attribute__((ext_vector_type(16))) _Float16 v16h;
typedef __attribute__((ext_vector_type(8)))  float    v8f;

#define WMMA16(a, b, c) \
  __builtin_amdgcn_wmma_f32_16x16x32_f16(false, (a), false, (b), (short)0, (c), false, false)

static __device__ inline v8f vzero8() {
  v8f z;
#pragma unroll
  for (int i = 0; i < 8; ++i) z[i] = 0.0f;
  return z;
}

// ----------------- f32 [K,N] -> f16 transposed [N,K] (tiled) ----------------
__global__ __launch_bounds__(256) void convert_transpose_kernel(
    const float* __restrict__ w, _Float16* __restrict__ wT, int K, int N) {
  __shared__ _Float16 tile[32][33];
  const int k0 = blockIdx.y * 32, n0 = blockIdx.x * 32;
  const int tx = threadIdx.x & 31, ty = threadIdx.x >> 5;  // 32 x 8
#pragma unroll
  for (int r = 0; r < 32; r += 8)
    tile[ty + r][tx] = (_Float16)w[(size_t)(k0 + ty + r) * N + n0 + tx];
  __syncthreads();
#pragma unroll
  for (int r = 0; r < 32; r += 8)
    wT[(size_t)(n0 + ty + r) * K + k0 + tx] = tile[tx][ty + r];
}

// --------------------------- adaLN: silu(c)@W + b ---------------------------
// c: [8,1024], w: [1024,6144], out cmod: [8,6144] fp32. grid(24,8), 256 thr.
__global__ __launch_bounds__(256) void adaln_kernel(const float* __restrict__ c,
                                                    const float* __restrict__ w,
                                                    const float* __restrict__ bias,
                                                    float* __restrict__ cmod) {
  __shared__ float sc[1024];
  const int b = blockIdx.y;
  for (int i = threadIdx.x; i < 1024; i += 256) {
    float v = c[b * 1024 + i];
    sc[i] = v / (1.0f + __expf(-v));  // silu
  }
  __syncthreads();
  const int col = blockIdx.x * 256 + threadIdx.x;
  float acc = bias[col];
  for (int k = 0; k < 1024; ++k) acc += sc[k] * w[(size_t)k * 6144 + col];
  cmod[(size_t)b * 6144 + col] = acc;
}

// --------------------- LayerNorm + modulate -> f16 row ----------------------
__global__ __launch_bounds__(256) void ln_mod_kernel(const float* __restrict__ x,
                                                     const float* __restrict__ cmod,
                                                     _Float16* __restrict__ out,
                                                     int shift_ofs, int scale_ofs) {
  __shared__ float row[1024];
  __shared__ float sred[256];
  const int r = blockIdx.x;
  const int b = r >> 10;  // S = 1024 rows per batch
  const float* xr = x + (size_t)r * 1024;
  const int t = threadIdx.x;
  float s = 0.0f;
  for (int i = t; i < 1024; i += 256) { float v = xr[i]; row[i] = v; s += v; }
  sred[t] = s; __syncthreads();
  for (int o = 128; o > 0; o >>= 1) { if (t < o) sred[t] += sred[t + o]; __syncthreads(); }
  const float mean = sred[0] * (1.0f / 1024.0f);
  __syncthreads();
  float s2 = 0.0f;
  for (int i = t; i < 1024; i += 256) { float d = row[i] - mean; s2 += d * d; }
  sred[t] = s2; __syncthreads();
  for (int o = 128; o > 0; o >>= 1) { if (t < o) sred[t] += sred[t + o]; __syncthreads(); }
  const float rstd = rsqrtf(sred[0] * (1.0f / 1024.0f) + 1e-6f);
  const float* cm = cmod + (size_t)b * 6144;
  for (int i = t; i < 1024; i += 256) {
    float v = (row[i] - mean) * rstd;
    v = v * (1.0f + cm[scale_ofs + i]) + cm[shift_ofs + i];
    out[(size_t)r * 1024 + i] = (_Float16)v;
  }
}

// ------------------------------ WMMA GEMM ----------------------------------
// C[M,N] = A[M,K] * B[K,N] with B given TRANSPOSED (BT = [N][K] row-major).
// Tiles: 128x128x32, 256 threads = 8 waves (4x2), each wave 32x64.
// Software-pipelined global->reg->LDS staging.
// mode 0: store f16. mode 1: +bias, gelu(tanh), store f16.
// mode 2: out_f32 = resid + gate[b,col] * (C + bias).
__global__ __launch_bounds__(256) void gemm_wmma_kernel(
    const _Float16* __restrict__ A, const _Float16* __restrict__ BT,
    int M, int N, int K, int mode,
    const float* __restrict__ bias, const float* __restrict__ gate,
    const float* __restrict__ resid, float* __restrict__ outf,
    _Float16* __restrict__ outh) {
  __shared__ alignas(32) _Float16 sA[128 * 32];
  __shared__ alignas(32) _Float16 sBT[128 * 32];
  const int t = threadIdx.x;
  const int wave = t >> 5, L = t & 31;
  const int lm = L & 15, lq = L >> 4;
  const int wm = wave >> 1, wn = wave & 1;
  const int mb = blockIdx.y * 128, nb = blockIdx.x * 128;

  v8f acc[2][4];
#pragma unroll
  for (int mi = 0; mi < 2; ++mi)
#pragma unroll
    for (int ni = 0; ni < 4; ++ni) acc[mi][ni] = vzero8();

  // staging: each thread handles one 16-half chunk of one row in each tile
  const int srow = t >> 1, scs = (t & 1) * 16;
  const _Float16* Ag = A  + (size_t)(mb + srow) * K + scs;
  const _Float16* Bg = BT + (size_t)(nb + srow) * K + scs;
  uint4* as = (uint4*)(sA  + srow * 32 + scs);
  uint4* bs = (uint4*)(sBT + srow * 32 + scs);

  uint4 ra0 = ((const uint4*)Ag)[0], ra1 = ((const uint4*)Ag)[1];
  uint4 rb0 = ((const uint4*)Bg)[0], rb1 = ((const uint4*)Bg)[1];

  for (int k0 = 32;; k0 += 32) {
    as[0] = ra0; as[1] = ra1;
    bs[0] = rb0; bs[1] = rb1;
    __syncthreads();
    if (k0 < K) {  // prefetch next tile while computing this one
      ra0 = ((const uint4*)(Ag + k0))[0]; ra1 = ((const uint4*)(Ag + k0))[1];
      rb0 = ((const uint4*)(Bg + k0))[0]; rb1 = ((const uint4*)(Bg + k0))[1];
    }

    v16h af[2];
#pragma unroll
    for (int mi = 0; mi < 2; ++mi) {
      const int m = wm * 32 + mi * 16 + lm;
      const int kq = lq << 4;
#pragma unroll
      for (int h = 0; h < 16; ++h)
        af[mi][h] = sA[m * 32 + kq + h];
    }
#pragma unroll
    for (int ni = 0; ni < 4; ++ni) {
      const v16h bf =
          *(const v16h*)(sBT + (size_t)(wn * 64 + ni * 16 + lm) * 32 + (lq << 4));
      acc[0][ni] = WMMA16(af[0], bf, acc[0][ni]);
      acc[1][ni] = WMMA16(af[1], bf, acc[1][ni]);
    }
    __syncthreads();
    if (k0 >= K) break;
  }

#pragma unroll
  for (int mi = 0; mi < 2; ++mi)
#pragma unroll
    for (int ni = 0; ni < 4; ++ni)
#pragma unroll
      for (int r = 0; r < 8; ++r) {
        const int grow = mb + wm * 32 + mi * 16 + r + (lq << 3);
        const int gcol = nb + wn * 64 + ni * 16 + lm;
        float v = acc[mi][ni][r];
        if (mode == 0) {
          outh[(size_t)grow * N + gcol] = (_Float16)v;
        } else if (mode == 1) {
          v += bias[gcol];
          const float u = v + 0.044715f * v * v * v;
          v = 0.5f * v * (1.0f + tanhf(0.7978845608028654f * u));
          outh[(size_t)grow * N + gcol] = (_Float16)v;
        } else {
          v += bias[gcol];
          const int bidx = grow >> 10;  // S = 1024
          v = resid[(size_t)grow * N + gcol] + gate[(size_t)bidx * 6144 + gcol] * v;
          outf[(size_t)grow * N + gcol] = v;
        }
      }
}

// --------------------------- Flash attention -------------------------------
// qkv f16 [B,S,3H] with column order (k | q | v). grid(8, 16, 8): 128 q-rows
// per block (16 per wave). Online softmax over 32-row K/V tiles. V is staged
// transposed so P@V fragments are contiguous v16h LDS reads.
__global__ __launch_bounds__(256) void attn_kernel(const _Float16* __restrict__ qkv,
                                                   _Float16* __restrict__ y) {
  const int S = 1024, H = 1024, CH = 64;
  __shared__ alignas(32) _Float16 sK[32 * 64];    // [krow][c]
  __shared__ alignas(32) _Float16 sVT[64 * 32];   // [c][krow]  (transposed)
  __shared__ alignas(32) _Float16 sP[8 * 16 * 32];
  const int t = threadIdx.x, wave = t >> 5, L = t & 31;
  const int lm = L & 15, lq = L >> 4;
  const int head = blockIdx.y, b = blockIdx.z;
  const int qrow0 = blockIdx.x * 128 + wave * 16;
  const _Float16* qb = qkv + (size_t)b * S * 3072;
  _Float16* sPw = sP + wave * 512;

  // q fragments (A layout), scaled by 1/64; vectorized 16B global loads
  v16h aq[2];
  {
    const _Float16* qp = qb + (size_t)(qrow0 + lm) * 3072 + H + head * CH;
    const _Float16 qs = (_Float16)(1.0f / 64.0f);
#pragma unroll
    for (int kf = 0; kf < 2; ++kf) {
      const int kbase = kf * 32 + (lq << 4);
      union { uint4 u; _Float16 h[8]; } lo, hi;
      lo.u = *(const uint4*)(qp + kbase);
      hi.u = *(const uint4*)(qp + kbase + 8);
#pragma unroll
      for (int h = 0; h < 8; ++h) {
        aq[kf][h]     = lo.h[h] * qs;
        aq[kf][h + 8] = hi.h[h] * qs;
      }
    }
  }

  float mi[8], li[8];
  v8f yacc[4];
#pragma unroll
  for (int r = 0; r < 8; ++r) { mi[r] = -1e30f; li[r] = 0.0f; }
#pragma unroll
  for (int c0 = 0; c0 < 4; ++c0) yacc[c0] = vzero8();

  const int ldrow = t >> 3, ldcs = (t & 7) * 8;  // 8 halfs per thread
  for (int j = 0; j < S; j += 32) {
    *(uint4*)(sK + ldrow * 64 + ldcs) =
        *(const uint4*)(qb + (size_t)(j + ldrow) * 3072 + head * CH + ldcs);
    {  // V: load 8 halfs of row ldrow, scatter transposed into sVT
      union { uint4 u; _Float16 h[8]; } vv;
      vv.u = *(const uint4*)(qb + (size_t)(j + ldrow) * 3072 + 2048 + head * CH + ldcs);
#pragma unroll
      for (int i = 0; i < 8; ++i) sVT[(ldcs + i) * 32 + ldrow] = vv.h[i];
    }
    __syncthreads();

    // scores: 16 q-rows x 32 k-rows, K-dim = CH = 64 (two chained wmma)
    v8f sc[2];
#pragma unroll
    for (int ns = 0; ns < 2; ++ns) {
      v8f z = vzero8();
      const int krow = ns * 16 + lm;  // B-frag lane = N = k-row
#pragma unroll
      for (int kf = 0; kf < 2; ++kf) {
        const v16h bk = *(const v16h*)(sK + krow * 64 + kf * 32 + (lq << 4));
        z = WMMA16(aq[kf], bk, z);
      }
      sc[ns] = z;
    }

    // online softmax (rows live in 16-lane groups; reduce with xor 1..8)
#pragma unroll
    for (int r = 0; r < 8; ++r) {
      float tm = fmaxf(sc[0][r], sc[1][r]);
#pragma unroll
      for (int o = 8; o >= 1; o >>= 1) tm = fmaxf(tm, __shfl_xor(tm, o, 32));
      const float mnew = fmaxf(mi[r], tm);
      const float alpha = __expf(mi[r] - mnew);
      const float p0 = __expf(sc[0][r] - mnew);
      const float p1 = __expf(sc[1][r] - mnew);
      float s = p0 + p1;
#pragma unroll
      for (int o = 8; o >= 1; o >>= 1) s += __shfl_xor(s, o, 32);
      li[r] = li[r] * alpha + s;
      mi[r] = mnew;
#pragma unroll
      for (int c0 = 0; c0 < 4; ++c0) yacc[c0][r] *= alpha;
      const int m = r + (lq << 3);
      sPw[m * 32 + lm]      = (_Float16)p0;
      sPw[m * 32 + 16 + lm] = (_Float16)p1;
    }
    __syncthreads();

    // y += P(16x32) @ V(32x64); V fragments contiguous in sVT
    v16h pf;
    {
      const int kq = lq << 4;
#pragma unroll
      for (int h = 0; h < 16; ++h)
        pf[h] = sPw[lm * 32 + kq + h];
    }
#pragma unroll
    for (int c0 = 0; c0 < 4; ++c0) {
      const v16h bv =
          *(const v16h*)(sVT + (size_t)(c0 * 16 + lm) * 32 + (lq << 4));
      yacc[c0] = WMMA16(pf, bv, yacc[c0]);
    }
    __syncthreads();
  }

#pragma unroll
  for (int c0 = 0; c0 < 4; ++c0)
#pragma unroll
    for (int r = 0; r < 8; ++r) {
      const int m = r + (lq << 3);
      const float v = yacc[c0][r] / li[r];
      y[(size_t)b * S * H + (size_t)(qrow0 + m) * H + head * CH + c0 * 16 + lm] =
          (_Float16)v;
    }
}

// ------------------------------- launcher ----------------------------------
extern "C" void kernel_launch(void* const* d_in, const int* in_sizes, int n_in,
                              void* d_out, int out_size, void* d_ws, size_t ws_size,
                              hipStream_t stream) {
  (void)in_sizes; (void)n_in; (void)out_size; (void)ws_size;
  const float* x      = (const float*)d_in[0];
  const float* c      = (const float*)d_in[1];
  const float* w_ada  = (const float*)d_in[2];
  const float* b_ada  = (const float*)d_in[3];
  const float* w_qkv  = (const float*)d_in[4];
  const float* w_proj = (const float*)d_in[5];
  const float* b_proj = (const float*)d_in[6];
  const float* w_mlp1 = (const float*)d_in[7];
  const float* b_mlp1 = (const float*)d_in[8];
  const float* w_mlp2 = (const float*)d_in[9];
  const float* b_mlp2 = (const float*)d_in[10];
  float* out = (float*)d_out;

  char* ws = (char*)d_ws;
  size_t o = 0;
  auto alloc = [&](size_t bytes) {
    void* p = ws + o;
    o = (o + bytes + 255) & ~(size_t)255;
    return p;
  };
  float*    cmod   = (float*)   alloc((size_t)8 * 6144 * 4);
  _Float16* whqkvT = (_Float16*)alloc((size_t)1024 * 3072 * 2);  // [3072][1024]
  _Float16* whprojT= (_Float16*)alloc((size_t)1024 * 1024 * 2);  // [1024][1024]
  _Float16* whm1T  = (_Float16*)alloc((size_t)1024 * 4096 * 2);  // [4096][1024]
  _Float16* whm2T  = (_Float16*)alloc((size_t)4096 * 1024 * 2);  // [1024][4096]
  _Float16* xmod   = (_Float16*)alloc((size_t)8192 * 1024 * 2);
  _Float16* qkvh   = (_Float16*)alloc((size_t)8192 * 3072 * 2);
  _Float16* yh     = (_Float16*)alloc((size_t)8192 * 1024 * 2);
  float*    x1     = (float*)   alloc((size_t)8192 * 1024 * 4);
  _Float16* hbuf   = (_Float16*)alloc((size_t)8192 * 4096 * 2);

  // weights -> f16 transposed [N][K] (deterministic, done every call)
  convert_transpose_kernel<<<dim3(3072 / 32, 1024 / 32), 256, 0, stream>>>(
      w_qkv, whqkvT, 1024, 3072);
  convert_transpose_kernel<<<dim3(1024 / 32, 1024 / 32), 256, 0, stream>>>(
      w_proj, whprojT, 1024, 1024);
  convert_transpose_kernel<<<dim3(4096 / 32, 1024 / 32), 256, 0, stream>>>(
      w_mlp1, whm1T, 1024, 4096);
  convert_transpose_kernel<<<dim3(1024 / 32, 4096 / 32), 256, 0, stream>>>(
      w_mlp2, whm2T, 4096, 1024);

  // adaLN modulation vectors
  adaln_kernel<<<dim3(24, 8), 256, 0, stream>>>(c, w_ada, b_ada, cmod);

  // attention branch
  ln_mod_kernel<<<8192, 256, 0, stream>>>(x, cmod, xmod, 0 * 1024, 1 * 1024);
  gemm_wmma_kernel<<<dim3(3072 / 128, 8192 / 128), 256, 0, stream>>>(
      xmod, whqkvT, 8192, 3072, 1024, 0, nullptr, nullptr, nullptr, nullptr, qkvh);
  attn_kernel<<<dim3(8, 16, 8), 256, 0, stream>>>(qkvh, yh);
  gemm_wmma_kernel<<<dim3(1024 / 128, 8192 / 128), 256, 0, stream>>>(
      yh, whprojT, 8192, 1024, 1024, 2, b_proj, cmod + 2 * 1024, x, x1, nullptr);

  // MLP branch
  ln_mod_kernel<<<8192, 256, 0, stream>>>(x1, cmod, xmod, 3 * 1024, 4 * 1024);
  gemm_wmma_kernel<<<dim3(4096 / 128, 8192 / 128), 256, 0, stream>>>(
      xmod, whm1T, 8192, 4096, 1024, 1, b_mlp1, nullptr, nullptr, nullptr, hbuf);
  gemm_wmma_kernel<<<dim3(1024 / 128, 8192 / 128), 256, 0, stream>>>(
      hbuf, whm2T, 8192, 1024, 4096, 2, b_mlp2, cmod + 5 * 1024, x1, out, nullptr);
}
